// CombinedSurvLoss_5325759447810
// MI455X (gfx1250) — matrix operand order repaired and checked by reading, not compiled, not run on hardware
//
#include <hip/hip_runtime.h>
#include <hip/hip_bf16.h>

typedef __attribute__((ext_vector_type(16))) _Float16 v16h;
typedef __attribute__((ext_vector_type(8)))  float    v8f;

#define NROWS   8192
#define PANEL   2048
#define WPB     4      // waves (i-tiles) per block in rank kernel
#define EPSF    1e-7f

#if __has_builtin(__builtin_amdgcn_global_load_async_to_lds_b32) && \
    __has_builtin(__builtin_amdgcn_s_wait_asynccnt)
#define USE_ASYNC_STAGE 1
#else
#define USE_ASYNC_STAGE 0
#endif

#if USE_ASYNC_STAGE
// builtin signature (from hipcc diagnostics): (AS1 int*, AS3 int*, imm int, imm int)
#define AS_G1(p) ((__attribute__((address_space(1))) int*)(uintptr_t)(p))
#define AS_L3(p) ((__attribute__((address_space(3))) int*)(p))
#endif

// ---------------------------------------------------------------------------
// Kernel 1: per-row prep. hazards, survival cumprod, risk, exp(risk) split
// into f16 hi + f16 residual (so WMMA f16 products are exact: mask in {0,1}),
// and the NLL contribution with a deterministic block reduction.
// ---------------------------------------------------------------------------
__global__ void __launch_bounds__(256)
rowprep_kernel(const float* __restrict__ out4,
               const int*   __restrict__ y,
               const int*   __restrict__ c,
               float*       __restrict__ risk,
               _Float16*    __restrict__ ehi,
               _Float16*    __restrict__ elo,
               float*       __restrict__ nll_part)
{
    __shared__ float red[256];
    const int tid = threadIdx.x;
    const int i   = blockIdx.x * 256 + tid;

    float h0 = 1.0f / (1.0f + expf(-out4[i * 4 + 0]));
    float h1 = 1.0f / (1.0f + expf(-out4[i * 4 + 1]));
    float h2 = 1.0f / (1.0f + expf(-out4[i * 4 + 2]));
    float h3 = 1.0f / (1.0f + expf(-out4[i * 4 + 3]));

    float s0 = 1.0f - h0;
    float s1 = s0 * (1.0f - h1);
    float s2 = s1 * (1.0f - h2);
    float s3 = s2 * (1.0f - h3);

    float r = -(s0 + s1 + s2 + s3);          // risk in [-4, 0]
    risk[i] = r;

    float er = expf(r);                      // in [e^-4, 1]: no range issues
    _Float16 hi = (_Float16)er;
    _Float16 lo = (_Float16)(er - (float)hi);
    ehi[i] = hi;
    elo[i] = lo;

    // NLL (ALPHA == 0)
    int   yi = y[i];
    float sp = (yi == 0) ? 1.0f : (yi == 1) ? s0 : (yi == 2) ? s1 : s2; // S_padded[yi]
    float ht = (yi == 0) ? h0   : (yi == 1) ? h1 : (yi == 2) ? h2 : h3; // hazards[yi]
    float st = (yi == 0) ? s0   : (yi == 1) ? s1 : (yi == 2) ? s2 : s3; // S_padded[yi+1]
    float cf = (float)c[i];
    float tot = -cf * logf(fmaxf(st, EPSF))
              - (1.0f - cf) * (logf(fmaxf(sp, EPSF)) + logf(fmaxf(ht, EPSF)));

    red[tid] = tot;
    __syncthreads();
    #pragma unroll
    for (int off = 128; off > 0; off >>= 1) {
        if (tid < off) red[tid] += red[tid + off];
        __syncthreads();
    }
    if (tid == 0) nll_part[blockIdx.x] = red[0];
}

// ---------------------------------------------------------------------------
// Kernel 2: pairwise masked sums via WMMA.
// One wave owns a 16-row i-tile; loops over all j in 32-wide chunks.
//   A[16x32] f16 : mask(i,j) = (t[j] > t[i]) ? 1 : 0
//   B[32x16] f16 : col0 = e_hi[j], col1 = e_lo[j], col2 = 1.0, rest 0
//   C[16x16] f32 : col0+col1 = sum_j mask*exp(risk[j]),  col2 = count
//
// B operand is built branch-free: the LDS region sB holds
//   [hi panel (PANEL halves) | lo panel (PANEL halves) | ones16 | zeros16]
// as an aligned v16h array; each lane resolves a fixed base + stride once.
// ---------------------------------------------------------------------------
__global__ void __launch_bounds__(WPB * 32)
rank_wmma_kernel(const float*    __restrict__ t,
                 const int*      __restrict__ c,
                 const float*    __restrict__ risk,
                 const _Float16* __restrict__ ehi,
                 const _Float16* __restrict__ elo,
                 float*          __restrict__ per_part,
                 float*          __restrict__ cnt_part)
{
    __shared__ float sT[PANEL];
    __shared__ v16h  sB[(2 * PANEL + 32) / 16];   // hi | lo | ones | zeros
    __shared__ float cbuf[WPB][3][16];

    const int tid   = threadIdx.x;
    const int wave  = tid >> 5;
    const int lane  = tid & 31;
    const int halfg = lane >> 4;       // lane group 0/1
    const int mrow  = lane & 15;
    const int tile  = blockIdx.x * WPB + wave;
    const int i0    = tile * 16;

    const float t_i   = t[i0 + mrow];  // this lane's A-rows all share row M=mrow
    const int   ncol  = mrow;          // this lane's B column
    const int   koffA = halfg * 16;    // contiguous A layout: lanes>=16 hold K+16

    // Per-lane B source (v16h units): hi panel / lo panel / const ones / const zeros
    const int bbase16 = (ncol == 0) ? halfg
                      : (ncol == 1) ? (PANEL / 16 + halfg)
                      : (ncol == 2) ? (2 * PANEL / 16)
                      :               (2 * PANEL / 16 + 1);
    const int bsel = (ncol < 2) ? 1 : 0;

    // init constant ones/zeros vectors (once)
    if (tid < 16) {
        _Float16* sH = (_Float16*)sB;
        sH[2 * PANEL + tid]      = (_Float16)1.0f;
        sH[2 * PANEL + 16 + tid] = (_Float16)0.0f;
    }

    v8f acc = {0.f, 0.f, 0.f, 0.f, 0.f, 0.f, 0.f, 0.f};

    for (int pb = 0; pb < NROWS; pb += PANEL) {
        // ---- stage panel: t, e_hi, e_lo for j in [pb, pb+PANEL) ----
#if USE_ASYNC_STAGE
        {
            const unsigned* gT  = (const unsigned*)(t + pb);      // PANEL words
            const unsigned* gHi = (const unsigned*)(ehi + pb);    // PANEL/2 words
            const unsigned* gLo = (const unsigned*)(elo + pb);    // PANEL/2 words
            unsigned* lT  = (unsigned*)sT;
            unsigned* lHi = (unsigned*)sB;
            unsigned* lLo = (unsigned*)sB + PANEL / 2;
            for (int e = tid; e < PANEL; e += WPB * 32)
                __builtin_amdgcn_global_load_async_to_lds_b32(
                    AS_G1(gT + e), AS_L3(lT + e), 0, 0);
            for (int e = tid; e < PANEL / 2; e += WPB * 32) {
                __builtin_amdgcn_global_load_async_to_lds_b32(
                    AS_G1(gHi + e), AS_L3(lHi + e), 0, 0);
                __builtin_amdgcn_global_load_async_to_lds_b32(
                    AS_G1(gLo + e), AS_L3(lLo + e), 0, 0);
            }
            __builtin_amdgcn_s_wait_asynccnt(0);
        }
#else
        {
            _Float16* sH = (_Float16*)sB;
            for (int e = tid; e < PANEL; e += WPB * 32) {
                sT[e]          = t[pb + e];
                sH[e]          = ehi[pb + e];
                sH[PANEL + e]  = elo[pb + e];
            }
        }
#endif
        if (pb + PANEL < NROWS)
            __builtin_prefetch(&t[pb + PANEL + tid * 4], 0, 0);
        __syncthreads();

        for (int jb = 0; jb < PANEL; jb += 32) {
            v16h a;
            // A: contiguous k order, same as B — element e of lane maps to
            // K = e + 16*halfg
            #pragma unroll
            for (int e = 0; e < 16; ++e) {
                int k = e + koffA;
                a[e] = (sT[jb + k] > t_i) ? (_Float16)1.0f : (_Float16)0.0f;
            }
            // B: one aligned 32B vector load per lane, no branches
            v16h b = sB[bbase16 + (jb >> 4) * bsel];

            acc = __builtin_amdgcn_wmma_f32_16x16x32_f16(
                false, a, false, b, (short)0, acc, false, false);
        }
        __syncthreads();
    }

    // C/D layout: vgpr v, lanes 0-15 -> (M=v, N=lane); lanes 16-31 -> (M=v+8, N=lane-16)
    if (ncol < 3) {
        #pragma unroll
        for (int v = 0; v < 8; ++v)
            cbuf[wave][ncol][v + 8 * halfg] = acc[v];
    }
    __syncthreads();

    float per = 0.0f, vcnt = 0.0f;
    if (lane < 16) {
        int   i   = i0 + lane;
        float se  = cbuf[wave][0][lane] + cbuf[wave][1][lane]; // sum_j mask*e^risk
        float cnt = cbuf[wave][2][lane];                       // sum_j mask
        bool valid = (c[i] == 0) && (cnt > 0.5f);
        if (valid) {
            per  = logf(se) - risk[i];   // lse - risk[i]
            vcnt = 1.0f;
        }
    }
    #pragma unroll
    for (int off = 16; off > 0; off >>= 1) {
        per  += __shfl_down(per,  off, 32);
        vcnt += __shfl_down(vcnt, off, 32);
    }
    if (lane == 0) {
        per_part[tile] = per;
        cnt_part[tile] = vcnt;
    }
}

// ---------------------------------------------------------------------------
// Kernel 3: deterministic final reduction -> scalar loss.
// ---------------------------------------------------------------------------
__global__ void __launch_bounds__(512)
finalize_kernel(const float* __restrict__ nll_part,
                const float* __restrict__ per_part,
                const float* __restrict__ cnt_part,
                float*       __restrict__ out)
{
    __shared__ float rp[512];
    __shared__ float rc[512];
    const int tid = threadIdx.x;
    rp[tid] = per_part[tid];
    rc[tid] = cnt_part[tid];
    __syncthreads();
    #pragma unroll
    for (int off = 256; off > 0; off >>= 1) {
        if (tid < off) { rp[tid] += rp[tid + off]; rc[tid] += rc[tid + off]; }
        __syncthreads();
    }
    if (tid == 0) {
        float nll = 0.0f;
        for (int i = 0; i < 32; ++i) nll += nll_part[i];   // fixed order
        nll /= (float)NROWS;
        float rank = (rc[0] > 0.0f) ? (rp[0] / fmaxf(rc[0], 1.0f)) : 0.0f;
        out[0] = nll + 0.5f * rank;
    }
}

// ---------------------------------------------------------------------------
extern "C" void kernel_launch(void* const* d_in, const int* in_sizes, int n_in,
                              void* d_out, int out_size, void* d_ws, size_t ws_size,
                              hipStream_t stream)
{
    const float* outputs = (const float*)d_in[0];
    const int*   y       = (const int*)d_in[1];
    const float* t       = (const float*)d_in[2];
    const int*   c       = (const int*)d_in[3];

    char* ws = (char*)d_ws;
    float*    risk     = (float*)(ws + 0);              // 8192 f32 = 32 KB
    _Float16* ehi      = (_Float16*)(ws + 32768);       // 8192 f16 = 16 KB
    _Float16* elo      = (_Float16*)(ws + 49152);       // 8192 f16 = 16 KB
    float*    nll_part = (float*)(ws + 65536);          // 32 f32
    float*    per_part = (float*)(ws + 65536 + 128);    // 512 f32
    float*    cnt_part = per_part + 512;                // 512 f32

    rowprep_kernel<<<NROWS / 256, 256, 0, stream>>>(outputs, y, c, risk, ehi, elo, nll_part);

    const int tiles = NROWS / 16;                       // 512 i-tiles
    rank_wmma_kernel<<<tiles / WPB, WPB * 32, 0, stream>>>(t, c, risk, ehi, elo,
                                                           per_part, cnt_part);

    finalize_kernel<<<1, 512, 0, stream>>>(nll_part, per_part, cnt_part, (float*)d_out);

    (void)in_sizes; (void)n_in; (void)out_size; (void)ws_size;
}
